// Block_40810779246681
// MI455X (gfx1250) — matrix operand order repaired and checked by reading, not compiled, not run on hardware
//
#include <hip/hip_runtime.h>
#include <hip/hip_bf16.h>
#include <math.h>

// ---------------- problem dims ----------------
static constexpr int Bn = 8, Sn = 512, Dn = 768, Hn = 12, En = 8, FFn = 3072;
static constexpr int HDn = 64;
static constexpr int Tn  = Bn * Sn;   // 4096 tokens
static constexpr int BHn = Bn * Hn;   // 96

// ---------------- types ----------------
typedef __attribute__((ext_vector_type(16))) _Float16 v16h;
typedef __attribute__((ext_vector_type(8)))  float    v8f;
typedef __attribute__((ext_vector_type(4)))  unsigned u32x4;
typedef __attribute__((ext_vector_type(8)))  int      i32x8;
typedef __attribute__((ext_vector_type(4)))  int      i32x4;

union FragU { v16h v; unsigned u[8]; };
union PackU { _Float16 h[2]; unsigned u; };

// ---------------- WMMA helpers (wave32 VGPR layouts per CDNA5 ISA) ----------------
// A fragment 16x32 f16 from row-major LDS tile [rows][ld]:
//   lane m=lane&15 owns row m; hi=lane>>4 selects K offset +8; VGPR v holds K pair
//   K(v,h) = (v&3)*2 + h + hi*8 + (v>>2)*16  (pairs contiguous -> one b32 ds load each)
__device__ inline v16h load_a_frag(const _Float16* lds, int row_base, int k_base, int ld) {
  const int lane = threadIdx.x & 31;
  const int m = lane & 15, hi = lane >> 4;
  const _Float16* p = lds + (size_t)(row_base + m) * ld + k_base;
  FragU f;
#pragma unroll
  for (int v = 0; v < 8; ++v) {
    const int k = ((v & 3) << 1) + (hi << 3) + ((v >> 2) << 4);
    f.u[v] = *(const unsigned*)(p + k);
  }
  return f.v;
}

// B fragment 32x16 f16 from N-major LDS tile [N][ld-of-K]:
//   lane n=lane&15 owns column n; hi=lane>>4 -> K offset +8; VGPR v holds K pair
//   K(v,h) = (v&3)*2 + h + hi*8 + (v>>2)*16 (same as A)
__device__ inline v16h load_b_frag(const _Float16* lds, int n_base, int k_base, int ld) {
  const int lane = threadIdx.x & 31;
  const int n = lane & 15, hi = lane >> 4;
  const _Float16* p = lds + (size_t)(n_base + n) * ld + k_base;
  FragU f;
#pragma unroll
  for (int v = 0; v < 8; ++v) f.u[v] = *(const unsigned*)(p + ((v & 3) << 1) + (hi << 3) + ((v >> 2) << 4));
  return f.v;
}

__device__ inline v8f wmma_f16(v16h a, v16h b, v8f c) {
  return __builtin_amdgcn_wmma_f32_16x16x32_f16(false, a, false, b, (short)0, c, false, false);
}

__device__ inline float geluf(float x) {
  return 0.5f * x * (1.0f + erff(x * 0.70710678118654752f));
}

// ---------------- async global -> LDS (ISA 15.18 op 98, ASYNCcnt-tracked DMA) ----------------
// VDST vgpr = LDS byte address; VADDR pair = 64-bit global address.
__device__ inline void async_g2l_b128(void* lds_dst, const void* gaddr) {
  const unsigned ldso = (unsigned)(unsigned long long)lds_dst;  // addr[31:0] = LDS offset
  asm volatile("global_load_async_to_lds_b128 %0, %1, off"
               :: "v"(ldso), "v"((unsigned long long)gaddr) : "memory");
}
__device__ inline void wait_async0() {
  asm volatile("s_wait_asynccnt 0" ::: "memory");
}

// ---------------- Tensor Data Mover: 2D f32 tile global -> LDS ----------------
// D# per CDNA5 ISA (group0: count/lds_addr/global_addr/type=2,
// group1: data_size, tensor dims, tile dims, dim0 stride). 6-arg builtin on this toolchain.
__device__ inline void tdm_load_2d_f32(void* lds_dst, const void* gsrc,
                                       unsigned tile_w, unsigned tile_h,
                                       unsigned row_stride_elems) {
  const unsigned long long ga = (unsigned long long)gsrc;
  const unsigned ldso = (unsigned)(unsigned long long)lds_dst;
  u32x4 g0;
  g0[0] = 1u;                                   // count=1, user mode, no gather
  g0[1] = ldso;                                 // lds_addr
  g0[2] = (unsigned)(ga & 0xffffffffu);         // global_addr[31:0]
  g0[3] = (unsigned)((ga >> 32) & 0x01ffffffu) | 0x80000000u;  // addr[56:32] | type=2
  const unsigned tw = row_stride_elems;         // tensor_dim0 (OOB guard: full row)
  const unsigned th = tile_h;                   // tensor_dim1
  i32x8 g1;
  g1[0] = (int)(2u << 16);                      // data_size=2 (4 bytes), mask=0
  g1[1] = (int)((tw & 0xffffu) << 16);          // tensor_dim0[15:0]
  g1[2] = (int)((tw >> 16) | ((th & 0xffffu) << 16));
  g1[3] = (int)((th >> 16) | (tile_w << 16));   // tile_dim0
  g1[4] = (int)(tile_h & 0xffffu);              // tile_dim1 (tile_dim2=0)
  g1[5] = (int)row_stride_elems;                // tensor_dim0_stride[31:0]
  g1[6] = 0;
  g1[7] = 0;
  i32x4 gz4 = {0, 0, 0, 0};
  i32x8 gz8 = {0, 0, 0, 0, 0, 0, 0, 0};
  __builtin_amdgcn_tensor_load_to_lds(g0, g1, gz4, gz4, gz8, 0);
}

// ---------------- staging helpers ----------------
// A tile: 64 rows x 32 f16 via async DMA; one b128 per thread (256 threads)
__device__ inline void stage_a_async(_Float16* dst, const _Float16* src, int row0,
                                     int ld_src, int kBase, int tid) {
  const int r = tid >> 2, c = (tid & 3) << 3;
  async_g2l_b128(&dst[r * 32 + c], &src[(size_t)(row0 + r) * ld_src + kBase + c]);
}
// B tile: f32 [32][128] global -> f16 N-major LDS [128][32].
// Each thread: K-pair (2 rows) x 8 N; batch 4x b128 loads, pack K pairs, 8x ds_store_b32.
__device__ inline void stage_b_f32(_Float16* dst, const float* w, int ldw,
                                   int kBase, int nBase, int tid) {
  const int k = (tid >> 4) << 1;        // even k (16 pairs)
  const int n0 = (tid & 15) << 3;       // 8 cols
  const float* s0 = w + (size_t)(kBase + k) * ldw + nBase + n0;
  const float* s1 = s0 + ldw;
  float a[8], b[8];
  *(float4*)&a[0] = *(const float4*)s0;
  *(float4*)&a[4] = *(const float4*)(s0 + 4);
  *(float4*)&b[0] = *(const float4*)s1;
  *(float4*)&b[4] = *(const float4*)(s1 + 4);
#pragma unroll
  for (int j = 0; j < 8; ++j) {
    PackU pk;
    pk.h[0] = (_Float16)a[j];
    pk.h[1] = (_Float16)b[j];
    *(unsigned*)&dst[(n0 + j) * 32 + k] = pk.u;
  }
}
// TDM-landed f32 [32][128] in LDS -> f16 N-major LDS [128][32] (same packing)
__device__ inline void convert_braw(_Float16* dst, const float* braw, int tid) {
  const int k = (tid >> 4) << 1;
  const int n0 = (tid & 15) << 3;
  float a[8], b[8];
#pragma unroll
  for (int j = 0; j < 8; ++j) {
    a[j] = braw[k * 128 + n0 + j];
    b[j] = braw[(k + 1) * 128 + n0 + j];
  }
#pragma unroll
  for (int j = 0; j < 8; ++j) {
    PackU pk;
    pk.h[0] = (_Float16)a[j];
    pk.h[1] = (_Float16)b[j];
    *(unsigned*)&dst[(n0 + j) * 32 + k] = pk.u;
  }
}

// ============================================================================
// 1) LayerNorm f32 -> f16 row (one block per token)
// ============================================================================
__global__ __launch_bounds__(256) void ln_kernel(const float* __restrict__ x,
                                                 const float* __restrict__ g,
                                                 const float* __restrict__ b,
                                                 _Float16* __restrict__ out) {
  __shared__ float s1[256], s2[256];
  const int t = blockIdx.x, tid = threadIdx.x;
  const float* row = x + (size_t)t * Dn;
  float sum = 0.f, ss = 0.f;
  for (int i = tid; i < Dn; i += 256) { float v = row[i]; sum += v; ss += v * v; }
  s1[tid] = sum; s2[tid] = ss;
  __syncthreads();
  for (int o = 128; o > 0; o >>= 1) {
    if (tid < o) { s1[tid] += s1[tid + o]; s2[tid] += s2[tid + o]; }
    __syncthreads();
  }
  const float mu = s1[0] * (1.0f / Dn);
  const float var = s2[0] * (1.0f / Dn) - mu * mu;
  const float r = rsqrtf(var + 1e-5f);
  for (int i = tid; i < Dn; i += 256)
    out[(size_t)t * Dn + i] = (_Float16)((row[i] - mu) * r * g[i] + b[i]);
}

// ============================================================================
// 2) QKV GEMM: h16[T,768] x qkv_w[768,2304] -> q[S,HD] (scaled), k[S,HD], vT[HD,S]
//    grid (64,18), block 256 (8 waves), double-buffered LDS, async A staging
// ============================================================================
__global__ __launch_bounds__(256) void gemm_qkv_kernel(const _Float16* __restrict__ h16,
                                                       const float* __restrict__ qkv_w,
                                                       _Float16* __restrict__ q16,
                                                       _Float16* __restrict__ k16,
                                                       _Float16* __restrict__ vT16) {
  __shared__ _Float16 As[2][64 * 32];
  __shared__ _Float16 Bs[2][128 * 32];
  const int tid = threadIdx.x;
  const int wave = tid >> 5, lane = tid & 31;
  const int wr = wave >> 1, wc = wave & 1;
  const int nIdx = lane & 15, hi = lane >> 4;
  const int rowBase = blockIdx.x * 64, colBase = blockIdx.y * 128;
  v8f acc[4] = {};
  stage_a_async(As[0], h16, rowBase, Dn, 0, tid);
  stage_b_f32(Bs[0], qkv_w, 3 * Dn, 0, colBase, tid);
  wait_async0();
  __syncthreads();
  int buf = 0;
  for (int kt = 0; kt < 24; ++kt) {
    if (kt + 2 < 24)  // warm L2 ahead -> global_prefetch_b8
      __builtin_prefetch(&h16[(size_t)(rowBase + (tid >> 2)) * Dn + (kt + 2) * 32], 0, 1);
    if (kt + 1 < 24) {
      stage_a_async(As[buf ^ 1], h16, rowBase, Dn, (kt + 1) * 32, tid);
      stage_b_f32(Bs[buf ^ 1], qkv_w, 3 * Dn, (kt + 1) * 32, colBase, tid);
    }
    const v16h a = load_a_frag(As[buf], wr * 16, 0, 32);
#pragma unroll
    for (int nt = 0; nt < 4; ++nt) {
      const v16h b = load_b_frag(Bs[buf], wc * 64 + nt * 16, 0, 32);
      acc[nt] = wmma_f16(a, b, acc[nt]);
    }
    wait_async0();
    __syncthreads();
    buf ^= 1;
  }
  // epilogue: scatter into q/k/vT head layouts; fold 1/sqrt(HD) into q
#pragma unroll
  for (int nt = 0; nt < 4; ++nt)
#pragma unroll
    for (int r = 0; r < 8; ++r) {
      const int row = rowBase + wr * 16 + r + hi * 8;       // token
      const int col = colBase + wc * 64 + nt * 16 + nIdx;   // 0..2303
      const int c = col / Dn, rem = col % Dn;
      const int hh = rem >> 6, d = rem & 63;
      const int bb = row >> 9, s = row & 511;
      const size_t bh = (size_t)(bb * Hn + hh);
      const float v = acc[nt][r];
      if (c == 0)      q16[(bh * Sn + s) * HDn + d] = (_Float16)(v * 0.125f);
      else if (c == 1) k16[(bh * Sn + s) * HDn + d] = (_Float16)v;
      else             vT16[(bh * HDn + d) * Sn + s] = (_Float16)v;
    }
}

// ============================================================================
// 3) Flash attention: per (qblock, b*h) WG of 128 threads (4 waves x 16 q rows)
//    K/V tiles staged by async DMA
// ============================================================================
__global__ __launch_bounds__(128) void attn_kernel(const _Float16* __restrict__ q16,
                                                   const _Float16* __restrict__ k16,
                                                   const _Float16* __restrict__ vT16,
                                                   _Float16* __restrict__ o16) {
  __shared__ _Float16 Qs[64 * 64];        // [q][d]
  __shared__ _Float16 Ks[64 * 64];        // [key][d]   (N-major for scores B)
  __shared__ _Float16 Vs[64 * 64];        // [d][key]   (N-major for o B)
  __shared__ _Float16 Ps[4][16 * 64];     // per-wave P (C-layout -> A-layout relay)
  const int tid = threadIdx.x;
  const int wave = tid >> 5, lane = tid & 31;
  const int nIdx = lane & 15, hi = lane >> 4;
  const int qb = blockIdx.x, bh = blockIdx.y;
  const int b = bh / Hn, h = bh % Hn;

  {  // stage Q (contiguous 8KB) via async DMA
    const _Float16* src = q16 + ((size_t)bh * Sn + qb * 64) * HDn;
#pragma unroll
    for (int i = 0; i < 4; ++i) {
      const int c = tid + i * 128;
      async_g2l_b128(&Qs[c * 8], &src[c * 8]);
    }
  }
  wait_async0();
  __syncthreads();
  const v16h qa0 = load_a_frag(Qs, wave * 16, 0, 64);
  const v16h qa1 = load_a_frag(Qs, wave * 16, 32, 64);

  float mrow[8], lrow[8];
#pragma unroll
  for (int r = 0; r < 8; ++r) { mrow[r] = -1e30f; lrow[r] = 0.f; }
  v8f acc[4] = {};

  for (int kb = 0; kb < Sn / 64; ++kb) {
    __syncthreads();
    {
      const _Float16* ksrc = k16 + ((size_t)bh * Sn + kb * 64) * HDn;
#pragma unroll
      for (int i = 0; i < 4; ++i) {
        const int c = tid + i * 128;
        async_g2l_b128(&Ks[c * 8], &ksrc[c * 8]);
      }
#pragma unroll
      for (int i = 0; i < 4; ++i) {
        const int c = tid + i * 128;
        const int d = c >> 3, part = (c & 7) * 8;
        async_g2l_b128(&Vs[d * 64 + part],
                       &vT16[((size_t)bh * HDn + d) * Sn + kb * 64 + part]);
      }
    }
    wait_async0();
    __syncthreads();

    v8f st[4];
#pragma unroll
    for (int nt = 0; nt < 4; ++nt) {
      v8f z = {};
      z = wmma_f16(qa0, load_b_frag(Ks, nt * 16, 0, 64), z);
      z = wmma_f16(qa1, load_b_frag(Ks, nt * 16, 32, 64), z);
      st[nt] = z;
    }
    // streaming softmax: row stats live per (r, lane-half); reduce over 16 lanes
    float cur[8];
#pragma unroll
    for (int r = 0; r < 8; ++r)
      cur[r] = fmaxf(fmaxf(st[0][r], st[1][r]), fmaxf(st[2][r], st[3][r]));
#pragma unroll
    for (int mk = 1; mk < 16; mk <<= 1)
#pragma unroll
      for (int r = 0; r < 8; ++r) cur[r] = fmaxf(cur[r], __shfl_xor(cur[r], mk, 32));
    float sc[8], rsum[8];
#pragma unroll
    for (int r = 0; r < 8; ++r) {
      const float mn = fmaxf(mrow[r], cur[r]);
      sc[r] = expf(mrow[r] - mn);
      mrow[r] = mn;
      rsum[r] = 0.f;
    }
#pragma unroll
    for (int nt = 0; nt < 4; ++nt)
#pragma unroll
      for (int r = 0; r < 8; ++r) {
        const float p = expf(st[nt][r] - mrow[r]);
        rsum[r] += p;
        Ps[wave][(r + hi * 8) * 64 + nt * 16 + nIdx] = (_Float16)p;
      }
#pragma unroll
    for (int mk = 1; mk < 16; mk <<= 1)
#pragma unroll
      for (int r = 0; r < 8; ++r) rsum[r] += __shfl_xor(rsum[r], mk, 32);
#pragma unroll
    for (int r = 0; r < 8; ++r) lrow[r] = lrow[r] * sc[r] + rsum[r];
#pragma unroll
    for (int dt = 0; dt < 4; ++dt)
#pragma unroll
      for (int r = 0; r < 8; ++r) acc[dt][r] = acc[dt][r] * sc[r];

    const v16h pa0 = load_a_frag(Ps[wave], 0, 0, 64);
    const v16h pa1 = load_a_frag(Ps[wave], 0, 32, 64);
#pragma unroll
    for (int dt = 0; dt < 4; ++dt) {
      acc[dt] = wmma_f16(pa0, load_b_frag(Vs, dt * 16, 0, 64), acc[dt]);
      acc[dt] = wmma_f16(pa1, load_b_frag(Vs, dt * 16, 32, 64), acc[dt]);
    }
  }
#pragma unroll
  for (int dt = 0; dt < 4; ++dt)
#pragma unroll
    for (int r = 0; r < 8; ++r) {
      const int srow = qb * 64 + wave * 16 + r + hi * 8;
      const int t = b * Sn + srow;
      const int col = h * HDn + dt * 16 + nIdx;
      o16[(size_t)t * Dn + col] = (_Float16)(acc[dt][r] / lrow[r]);
    }
}

// ============================================================================
// 4) proj GEMM + residual: xmid = x + o16 @ proj_w + proj_b   grid (64,6)
// ============================================================================
__global__ __launch_bounds__(256) void gemm_proj_kernel(const _Float16* __restrict__ o16,
                                                        const float* __restrict__ proj_w,
                                                        const float* __restrict__ proj_b,
                                                        const float* __restrict__ x,
                                                        float* __restrict__ xmid) {
  __shared__ _Float16 As[2][64 * 32];
  __shared__ _Float16 Bs[2][128 * 32];
  const int tid = threadIdx.x;
  const int wave = tid >> 5, lane = tid & 31;
  const int wr = wave >> 1, wc = wave & 1;
  const int nIdx = lane & 15, hi = lane >> 4;
  const int rowBase = blockIdx.x * 64, colBase = blockIdx.y * 128;
  v8f acc[4] = {};
  stage_a_async(As[0], o16, rowBase, Dn, 0, tid);
  stage_b_f32(Bs[0], proj_w, Dn, 0, colBase, tid);
  wait_async0();
  __syncthreads();
  int buf = 0;
  for (int kt = 0; kt < 24; ++kt) {
    if (kt + 1 < 24) {
      stage_a_async(As[buf ^ 1], o16, rowBase, Dn, (kt + 1) * 32, tid);
      stage_b_f32(Bs[buf ^ 1], proj_w, Dn, (kt + 1) * 32, colBase, tid);
    }
    const v16h a = load_a_frag(As[buf], wr * 16, 0, 32);
#pragma unroll
    for (int nt = 0; nt < 4; ++nt)
      acc[nt] = wmma_f16(a, load_b_frag(Bs[buf], wc * 64 + nt * 16, 0, 32), acc[nt]);
    wait_async0();
    __syncthreads();
    buf ^= 1;
  }
#pragma unroll
  for (int nt = 0; nt < 4; ++nt)
#pragma unroll
    for (int r = 0; r < 8; ++r) {
      const int row = rowBase + wr * 16 + r + hi * 8;
      const int col = colBase + wc * 64 + nt * 16 + nIdx;
      xmid[(size_t)row * Dn + col] =
          acc[nt][r] + proj_b[col] + x[(size_t)row * Dn + col];
    }
}

// ============================================================================
// 5) router: logits argmax (softmax is monotonic)
// ============================================================================
__global__ __launch_bounds__(256) void router_kernel(const _Float16* __restrict__ h2,
                                                     const float* __restrict__ sw,
                                                     const float* __restrict__ sb,
                                                     int* __restrict__ routes) {
  const int t = blockIdx.x * blockDim.x + threadIdx.x;
  if (t >= Tn) return;
  float lg[En];
#pragma unroll
  for (int e = 0; e < En; ++e) lg[e] = sb[e];
  const _Float16* row = h2 + (size_t)t * Dn;
  for (int d = 0; d < Dn; ++d) {
    const float hv = (float)row[d];
    const float* w = sw + (size_t)d * En;
#pragma unroll
    for (int e = 0; e < En; ++e) lg[e] += hv * w[e];
  }
  int best = 0;
  float bv = lg[0];
#pragma unroll
  for (int e = 1; e < En; ++e)
    if (lg[e] > bv) { bv = lg[e]; best = e; }
  routes[t] = best;
}

// ============================================================================
// 6) counting-sort tokens by expert (tiny; deterministic scatter by thread 0)
// ============================================================================
__global__ __launch_bounds__(256) void route_build_kernel(const int* __restrict__ routes,
                                                          int* __restrict__ perm,
                                                          int* __restrict__ cnt,
                                                          int* __restrict__ offs) {
  __shared__ int c[En];
  if (threadIdx.x < En) c[threadIdx.x] = 0;
  __syncthreads();
  for (int t = threadIdx.x; t < Tn; t += 256) atomicAdd(&c[routes[t]], 1);
  __syncthreads();
  if (threadIdx.x == 0) {
    int o = 0, cur[En];
    for (int e = 0; e < En; ++e) { offs[e] = o; cnt[e] = c[e]; cur[e] = o; o += c[e]; }
    for (int t = 0; t < Tn; ++t) { const int e = routes[t]; perm[cur[e]++] = t; }
  }
}

// ============================================================================
// 7) MoE GEMM1: mid[sorted,FF] = gelu(h2[perm] @ w1[e] + b1[e])
//    grid (24, 64, 8); B tile via Tensor Data Mover, A tile via async DMA
// ============================================================================
__global__ __launch_bounds__(256) void gemm_moe1_kernel(const _Float16* __restrict__ h2,
                                                        const float* __restrict__ w1,
                                                        const float* __restrict__ b1,
                                                        const int* __restrict__ perm,
                                                        const int* __restrict__ cnt,
                                                        const int* __restrict__ offs,
                                                        _Float16* __restrict__ mid) {
  const int e = blockIdx.z, tileRow = blockIdx.y, nb = blockIdx.x * 128;
  const int cn = cnt[e];
  if (tileRow * 64 >= cn) return;
  const int base = offs[e];
  __shared__ _Float16 As[64 * 32];
  __shared__ float    Braw[32 * 128];
  __shared__ _Float16 Bs[128 * 32];
  __shared__ int rowTok[64];
  const int tid = threadIdx.x;
  const int wave = tid >> 5, lane = tid & 31;
  const int wr = wave >> 1, wc = wave & 1;
  const int nIdx = lane & 15, hi = lane >> 4;
  if (tid < 64) {
    int sr = tileRow * 64 + tid;
    if (sr >= cn) sr = cn - 1;
    rowTok[tid] = perm[base + sr];
  }
  __syncthreads();
  v8f acc[4] = {};
  for (int kt = 0; kt < 24; ++kt) {
    const int kBase = kt * 32;
    if (tid < 32) {  // wave 0 drives the TDM
      tdm_load_2d_f32(Braw, w1 + ((size_t)e * Dn + kBase) * FFn + nb, 128, 32, FFn);
      __builtin_amdgcn_s_wait_tensorcnt(0);
    }
    {  // gathered A staging (async DMA) in parallel with TDM
      const int r = tid >> 2, c = (tid & 3) << 3;
      async_g2l_b128(&As[r * 32 + c], &h2[(size_t)rowTok[r] * Dn + kBase + c]);
    }
    wait_async0();
    __syncthreads();
    convert_braw(Bs, Braw, tid);  // f32 -> f16 N-major, packed b32 stores
    __syncthreads();
    const v16h a = load_a_frag(As, wr * 16, 0, 32);
#pragma unroll
    for (int nt = 0; nt < 4; ++nt)
      acc[nt] = wmma_f16(a, load_b_frag(Bs, wc * 64 + nt * 16, 0, 32), acc[nt]);
    __syncthreads();
  }
#pragma unroll
  for (int nt = 0; nt < 4; ++nt)
#pragma unroll
    for (int r = 0; r < 8; ++r) {
      const int M = wr * 16 + r + hi * 8;
      const int srow = tileRow * 64 + M;
      if (srow < cn) {
        const int col = nb + wc * 64 + nt * 16 + nIdx;
        mid[(size_t)(base + srow) * FFn + col] =
            (_Float16)geluf(acc[nt][r] + b1[(size_t)e * FFn + col]);
      }
    }
}

// ============================================================================
// 8) MoE GEMM2 + scatter + residual: out[tok] = xmid[tok] + gelu(mid @ w2[e] + b2[e])
//    grid (6, 64, 8)
// ============================================================================
__global__ __launch_bounds__(256) void gemm_moe2_kernel(const _Float16* __restrict__ mid,
                                                        const float* __restrict__ w2,
                                                        const float* __restrict__ b2,
                                                        const int* __restrict__ perm,
                                                        const int* __restrict__ cnt,
                                                        const int* __restrict__ offs,
                                                        const float* __restrict__ xmid,
                                                        float* __restrict__ out) {
  const int e = blockIdx.z, tileRow = blockIdx.y, nb = blockIdx.x * 128;
  const int cn = cnt[e];
  if (tileRow * 64 >= cn) return;
  const int base = offs[e];
  __shared__ _Float16 As[64 * 32];
  __shared__ float    Braw[32 * 128];
  __shared__ _Float16 Bs[128 * 32];
  const int tid = threadIdx.x;
  const int wave = tid >> 5, lane = tid & 31;
  const int wr = wave >> 1, wc = wave & 1;
  const int nIdx = lane & 15, hi = lane >> 4;
  v8f acc[4] = {};
  for (int kt = 0; kt < 96; ++kt) {
    const int kBase = kt * 32;
    if (tid < 32) {
      tdm_load_2d_f32(Braw, w2 + ((size_t)e * FFn + kBase) * Dn + nb, 128, 32, Dn);
      __builtin_amdgcn_s_wait_tensorcnt(0);
    }
    {
      const int r = tid >> 2, c = (tid & 3) << 3;
      int sr = tileRow * 64 + r;
      if (sr >= cn) sr = cn - 1;
      async_g2l_b128(&As[r * 32 + c], &mid[(size_t)(base + sr) * FFn + kBase + c]);
    }
    wait_async0();
    __syncthreads();
    convert_braw(Bs, Braw, tid);
    __syncthreads();
    const v16h a = load_a_frag(As, wr * 16, 0, 32);
#pragma unroll
    for (int nt = 0; nt < 4; ++nt)
      acc[nt] = wmma_f16(a, load_b_frag(Bs, wc * 64 + nt * 16, 0, 32), acc[nt]);
    __syncthreads();
  }
#pragma unroll
  for (int nt = 0; nt < 4; ++nt)
#pragma unroll
    for (int r = 0; r < 8; ++r) {
      const int M = wr * 16 + r + hi * 8;
      const int srow = tileRow * 64 + M;
      if (srow < cn) {
        const int tok = perm[base + srow];
        const int col = nb + wc * 64 + nt * 16 + nIdx;
        out[(size_t)tok * Dn + col] =
            xmid[(size_t)tok * Dn + col] +
            geluf(acc[nt][r] + b2[(size_t)e * Dn + col]);
      }
    }
}

// ============================================================================
// host launcher
// ============================================================================
extern "C" void kernel_launch(void* const* d_in, const int* in_sizes, int n_in,
                              void* d_out, int out_size, void* d_ws, size_t ws_size,
                              hipStream_t stream) {
  (void)in_sizes; (void)n_in; (void)out_size; (void)ws_size;
  const float* x      = (const float*)d_in[0];
  // d_in[1] = indexes_list (unused, as in reference)
  const float* ln1_g  = (const float*)d_in[2];
  const float* ln1_b  = (const float*)d_in[3];
  const float* qkv_w  = (const float*)d_in[4];
  const float* proj_w = (const float*)d_in[5];
  const float* proj_b = (const float*)d_in[6];
  const float* ln2_g  = (const float*)d_in[7];
  const float* ln2_b  = (const float*)d_in[8];
  const float* sw     = (const float*)d_in[9];
  const float* sb     = (const float*)d_in[10];
  const float* w1     = (const float*)d_in[11];
  const float* b1     = (const float*)d_in[12];
  const float* w2     = (const float*)d_in[13];
  const float* b2     = (const float*)d_in[14];
  float* out = (float*)d_out;

  char* ws = (char*)d_ws;
  size_t off = 0;
  auto alloc = [&](size_t bytes) -> void* {
    void* p = ws + off;
    off += (bytes + 255) & ~(size_t)255;
    return p;
  };
  _Float16* h16   = (_Float16*)alloc((size_t)Tn * Dn * 2);
  _Float16* q16   = (_Float16*)alloc((size_t)BHn * Sn * HDn * 2);
  _Float16* k16   = (_Float16*)alloc((size_t)BHn * Sn * HDn * 2);
  _Float16* vT16  = (_Float16*)alloc((size_t)BHn * HDn * Sn * 2);
  _Float16* o16   = (_Float16*)alloc((size_t)Tn * Dn * 2);
  float*    xmid  = (float*)alloc((size_t)Tn * Dn * 4);
  _Float16* h2_16 = (_Float16*)alloc((size_t)Tn * Dn * 2);
  _Float16* mid16 = (_Float16*)alloc((size_t)Tn * FFn * 2);
  int*      routes = (int*)alloc((size_t)Tn * 4);
  int*      perm   = (int*)alloc((size_t)Tn * 4);
  int*      cnt    = (int*)alloc(En * 4);
  int*      offs   = (int*)alloc(En * 4);

  ln_kernel<<<Tn, 256, 0, stream>>>(x, ln1_g, ln1_b, h16);
  gemm_qkv_kernel<<<dim3(Tn / 64, (3 * Dn) / 128), 256, 0, stream>>>(h16, qkv_w, q16, k16, vT16);
  attn_kernel<<<dim3(Sn / 64, BHn), 128, 0, stream>>>(q16, k16, vT16, o16);
  gemm_proj_kernel<<<dim3(Tn / 64, Dn / 128), 256, 0, stream>>>(o16, proj_w, proj_b, x, xmid);
  ln_kernel<<<Tn, 256, 0, stream>>>(xmid, ln2_g, ln2_b, h2_16);
  router_kernel<<<Tn / 256, 256, 0, stream>>>(h2_16, sw, sb, routes);
  route_build_kernel<<<1, 256, 0, stream>>>(routes, perm, cnt, offs);
  gemm_moe1_kernel<<<dim3(FFn / 128, Tn / 64, En), 256, 0, stream>>>(h2_16, w1, b1, perm, cnt, offs, mid16);
  gemm_moe2_kernel<<<dim3(Dn / 128, Tn / 64, En), 256, 0, stream>>>(mid16, w2, b2, perm, cnt, offs, xmid, out);
}
